// Model_36721970381006
// MI455X (gfx1250) — matrix operand order repaired and checked by reading, not compiled, not run on hardware
//
#include <hip/hip_runtime.h>
#include <stdint.h>

// ---------------------------------------------------------------------------
// Types for CDNA5 WMMA
// ---------------------------------------------------------------------------
typedef __attribute__((ext_vector_type(16))) __bf16 v16bf;
typedef __attribute__((ext_vector_type(8)))  float  v8f;
typedef __attribute__((ext_vector_type(4)))  unsigned int v4u;

#define BB   64      // batch
#define SS   512     // sequence length
#define FF   64      // features
#define HH   1024    // hidden
#define G4H  4096    // 4*H (gates)

// ---------------------------------------------------------------------------
// helpers
// ---------------------------------------------------------------------------
__device__ __forceinline__ unsigned short f2bf(float f) {
    union { float f; uint32_t u; } v; v.f = f;
    uint32_t r = v.u + 0x7FFFu + ((v.u >> 16) & 1u);   // round-to-nearest-even
    return (unsigned short)(r >> 16);
}

// load 16 bf16 (two 16-byte chunks) into a WMMA fragment register set
__device__ __forceinline__ v16bf load_frag(const unsigned short* p0,
                                           const unsigned short* p1) {
    union { v16bf v; v4u q[2]; } u;
    u.q[0] = *reinterpret_cast<const v4u*>(p0);
    u.q[1] = *reinterpret_cast<const v4u*>(p1);
    return u.v;
}

__device__ __forceinline__ float sigm(float x) { return 1.0f / (1.0f + __expf(-x)); }

// ---------------------------------------------------------------------------
// setup kernels
// ---------------------------------------------------------------------------
__global__ void k_f32_to_bf16(const float* __restrict__ src,
                              unsigned short* __restrict__ dst, int n) {
    int i = blockIdx.x * blockDim.x + threadIdx.x;
    if (i < n) dst[i] = f2bf(src[i]);
}

__global__ void k_bias_combine(const float* __restrict__ a,
                               const float* __restrict__ b,
                               float* __restrict__ o, int n) {
    int i = blockIdx.x * blockDim.x + threadIdx.x;
    if (i < n) o[i] = a[i] + b[i];
}

__global__ void k_zero(uint32_t* __restrict__ p, int nwords) {
    int i = blockIdx.x * blockDim.x + threadIdx.x;
    if (i < nwords) p[i] = 0u;
}

// Instance norm over time. One wave per (b,f); writes xn as [S][B][F] bf16.
__global__ void k_instnorm(const float* __restrict__ x,      // [B,S,F]
                           unsigned short* __restrict__ xn,  // [S,B,F] bf16
                           float* __restrict__ mean,         // [B*F]
                           float* __restrict__ stdev) {      // [B*F]
    int wave = (blockIdx.x * blockDim.x + threadIdx.x) >> 5;
    int lane = threadIdx.x & 31;
    if (wave >= BB * FF) return;
    int b = wave / FF, f = wave % FF;
    const float* xp = x + (size_t)b * SS * FF + f;
    float s = 0.0f, s2 = 0.0f;
    for (int t = lane; t < SS; t += 32) {
        float v = xp[(size_t)t * FF];
        s += v; s2 += v * v;
    }
    #pragma unroll
    for (int o = 16; o; o >>= 1) {
        s  += __shfl_xor(s,  o, 32);
        s2 += __shfl_xor(s2, o, 32);
    }
    float m   = s / (float)SS;
    float var = s2 / (float)SS - m * m;
    float sd  = sqrtf(var + 1e-5f);
    if (lane == 0) { mean[wave] = m; stdev[wave] = sd; }
    float inv = 1.0f / sd;
    for (int t = lane; t < SS; t += 32)
        xn[(size_t)t * BB * FF + (size_t)b * FF + f] = f2bf((xp[(size_t)t * FF] - m) * inv);
}

// ---------------------------------------------------------------------------
// Recurrent gate GEMM (split-K, deterministic partials):
//   partial[slot][b][4H] = A_chunk @ W_chunk^T
// slots [0,Sh)     : h-path  (Ah[B,Kh] x Wh[4H,Kh])
// slots [Sh,Sh+Sx) : x-path  (Ax[B,Kx] x Wx[4H,Kx])
// One wave per (slot, gate, u-tile); each wave computes 4 M-tiles (B=64).
// ---------------------------------------------------------------------------
__global__ void k_gates_gemm(const unsigned short* __restrict__ Ah,
                             const unsigned short* __restrict__ Wh, int Kh, int Sh,
                             const unsigned short* __restrict__ Ax,
                             const unsigned short* __restrict__ Wx, int Kx, int Sx,
                             float* __restrict__ partial) {
    int wave = (blockIdx.x * blockDim.x + threadIdx.x) >> 5;
    int lane = threadIdx.x & 31;
    int slot = wave >> 8;            // / 256  (4 gates * 64 u-tiles)
    int rem  = wave & 255;
    int gate = rem >> 6;
    int ut   = rem & 63;
    if (slot >= Sh + Sx) return;

    const unsigned short* A;
    const unsigned short* W;
    int K, k0, kn;
    if (slot < Sh) { A = Ah; W = Wh; K = Kh; int kc = Kh / Sh; k0 = slot * kc;        kn = kc; }
    else           { A = Ax; W = Wx; K = Kx; int kc = Kx / Sx; k0 = (slot - Sh) * kc; kn = kc; }

    int n  = lane & 15;
    int hi = lane >> 4;

    v8f acc[4];
    #pragma unroll
    for (int m = 0; m < 4; ++m)
        #pragma unroll
        for (int e = 0; e < 8; ++e) acc[m][e] = 0.0f;

    const size_t wrow = (size_t)(gate * HH + ut * 16 + n) * (size_t)K;
    for (int kt = 0; kt < (kn >> 5); ++kt) {
        int kk = k0 + kt * 32;
        // B fragment: row n of W, two 8-elem K chunks at hi*8 and hi*8+16
        const unsigned short* wp = W + wrow + kk + hi * 8;
        v16bf bfrag = load_frag(wp, wp + 16);
        #pragma unroll
        for (int m = 0; m < 4; ++m) {
            // A fragment: row (m*16+n), two 8-elem K chunks at hi*8 and hi*8+16
            const unsigned short* ap = A + (size_t)(m * 16 + n) * (size_t)K + kk + hi * 8;
            v16bf afrag = load_frag(ap, ap + 16);
            acc[m] = __builtin_amdgcn_wmma_f32_16x16x32_bf16(
                false, afrag, false, bfrag, (short)0, acc[m], false, false);
        }
    }

    // write partials: C/D layout  M = m*16 + r + 8*hi, N = n
    float* pb = partial + (size_t)slot * BB * G4H + (size_t)(gate * HH + ut * 16 + n);
    #pragma unroll
    for (int m = 0; m < 4; ++m)
        #pragma unroll
        for (int r = 0; r < 8; ++r) {
            int brow = m * 16 + r + 8 * hi;
            pb[(size_t)brow * G4H] = acc[m][r];
        }
}

// ---------------------------------------------------------------------------
// LSTM cell pointwise: reduce split-K partials, add bias, activations,
// update c (fp32, in place), write h (bf16) and optionally relu(h) (bf16).
// ---------------------------------------------------------------------------
__global__ void k_lstm_pointwise(const float* __restrict__ partial, int nslots,
                                 const float* __restrict__ bias,     // [4H]
                                 float* __restrict__ c,              // [B,H]
                                 unsigned short* __restrict__ h_out, // [B,H] bf16
                                 unsigned short* __restrict__ hrelu) {
    int idx = blockIdx.x * blockDim.x + threadIdx.x;   // b*H + u
    if (idx >= BB * HH) return;
    int b = idx >> 10, u = idx & (HH - 1);
    float gi = bias[u];
    float gf = bias[HH + u];
    float gg = bias[2 * HH + u];
    float go = bias[3 * HH + u];
    for (int s = 0; s < nslots; ++s) {
        const float* p = partial + (size_t)s * BB * G4H + (size_t)b * G4H + u;
        gi += p[0];
        gf += p[HH];
        gg += p[2 * HH];
        go += p[3 * HH];
    }
    float ig = sigm(gi), fg = sigm(gf), gt = tanhf(gg), og = sigm(go);
    float cn = fg * c[idx] + ig * gt;
    float hn = og * tanhf(cn);
    c[idx] = cn;
    h_out[idx] = f2bf(hn);
    if (hrelu) hrelu[idx] = f2bf(hn > 0.0f ? hn : 0.0f);
}

// ---------------------------------------------------------------------------
// Decoder FC: y = relu(h1) @ fcW^T + fcb; writes de-normalized output row
// (time-reversed) and bf16 feedback x for the next decoder step.
// 16 waves, one 16x16 tile each (M=64, N=64, K=1024).
// ---------------------------------------------------------------------------
__global__ void k_fc(const unsigned short* __restrict__ hrelu, // [B,H] bf16
                     const unsigned short* __restrict__ fcW,   // [F,H] bf16
                     const float* __restrict__ fcb,            // [F]
                     const float* __restrict__ mean,           // [B*F]
                     const float* __restrict__ stdev,          // [B*F]
                     unsigned short* __restrict__ xnext,       // [B,F] bf16
                     float* __restrict__ out,                  // [B,S,F]
                     int t) {
    int wave = (blockIdx.x * blockDim.x + threadIdx.x) >> 5;
    int lane = threadIdx.x & 31;
    if (wave >= 16) return;
    int mt = wave >> 2, nt = wave & 3;
    int n = lane & 15, hi = lane >> 4;

    v8f acc;
    #pragma unroll
    for (int e = 0; e < 8; ++e) acc[e] = 0.0f;

    const size_t wrow = (size_t)(nt * 16 + n) * HH;
    const size_t arow = (size_t)(mt * 16 + n) * HH;
    for (int kt = 0; kt < (HH >> 5); ++kt) {
        int kk = kt * 32;
        const unsigned short* wp = fcW   + wrow + kk + hi * 8;
        const unsigned short* ap = hrelu + arow + kk + hi * 8;
        v16bf bfrag = load_frag(wp, wp + 16);
        v16bf afrag = load_frag(ap, ap + 16);
        acc = __builtin_amdgcn_wmma_f32_16x16x32_bf16(
            false, afrag, false, bfrag, (short)0, acc, false, false);
    }

    int f = nt * 16 + n;
    float bv = fcb[f];
    int srow = SS - 1 - t;            // reverse-time output
    #pragma unroll
    for (int r = 0; r < 8; ++r) {
        int b = mt * 16 + r + 8 * hi;
        float y = acc[r] + bv;
        xnext[b * FF + f] = f2bf(y);
        out[(size_t)b * SS * FF + (size_t)srow * FF + f] =
            y * stdev[b * FF + f] + mean[b * FF + f];
    }
}

// ---------------------------------------------------------------------------
// host side
// ---------------------------------------------------------------------------
extern "C" void kernel_launch(void* const* d_in, const int* in_sizes, int n_in,
                              void* d_out, int out_size, void* d_ws, size_t ws_size,
                              hipStream_t stream) {
    (void)in_sizes; (void)n_in; (void)out_size; (void)ws_size;

    const float* x_enc    = (const float*)d_in[0];
    const float* enc_Wih0 = (const float*)d_in[4];
    const float* enc_Whh0 = (const float*)d_in[5];
    const float* enc_bih0 = (const float*)d_in[6];
    const float* enc_bhh0 = (const float*)d_in[7];
    const float* enc_Wih1 = (const float*)d_in[8];
    const float* enc_Whh1 = (const float*)d_in[9];
    const float* enc_bih1 = (const float*)d_in[10];
    const float* enc_bhh1 = (const float*)d_in[11];
    const float* dec_Wih0 = (const float*)d_in[12];
    const float* dec_Whh0 = (const float*)d_in[13];
    const float* dec_bih0 = (const float*)d_in[14];
    const float* dec_bhh0 = (const float*)d_in[15];
    const float* dec_Wih1 = (const float*)d_in[16];
    const float* dec_Whh1 = (const float*)d_in[17];
    const float* dec_bih1 = (const float*)d_in[18];
    const float* dec_bhh1 = (const float*)d_in[19];
    const float* fc_W     = (const float*)d_in[20];
    const float* fc_b     = (const float*)d_in[21];
    float* out = (float*)d_out;

    // ---- workspace carve-up (256B aligned bumps) ----
    char* ws = (char*)d_ws;
    size_t off = 0;
    auto alloc = [&](size_t bytes) -> void* {
        void* p = ws + off;
        off = (off + bytes + 255) & ~(size_t)255;
        return p;
    };
    unsigned short* wEWih0 = (unsigned short*)alloc((size_t)G4H * FF * 2);
    unsigned short* wEWhh0 = (unsigned short*)alloc((size_t)G4H * HH * 2);
    unsigned short* wEWih1 = (unsigned short*)alloc((size_t)G4H * HH * 2);
    unsigned short* wEWhh1 = (unsigned short*)alloc((size_t)G4H * HH * 2);
    unsigned short* wDWih0 = (unsigned short*)alloc((size_t)G4H * FF * 2);
    unsigned short* wDWhh0 = (unsigned short*)alloc((size_t)G4H * HH * 2);
    unsigned short* wDWih1 = (unsigned short*)alloc((size_t)G4H * HH * 2);
    unsigned short* wDWhh1 = (unsigned short*)alloc((size_t)G4H * HH * 2);
    unsigned short* wFc    = (unsigned short*)alloc((size_t)FF * HH * 2);
    unsigned short* xn     = (unsigned short*)alloc((size_t)SS * BB * FF * 2);
    unsigned short* hL0    = (unsigned short*)alloc((size_t)BB * HH * 2);
    unsigned short* hL1    = (unsigned short*)alloc((size_t)BB * HH * 2);
    unsigned short* hrelu  = (unsigned short*)alloc((size_t)BB * HH * 2);
    unsigned short* xprev  = (unsigned short*)alloc((size_t)BB * FF * 2);
    float* biasE0 = (float*)alloc((size_t)G4H * 4);
    float* biasE1 = (float*)alloc((size_t)G4H * 4);
    float* biasD0 = (float*)alloc((size_t)G4H * 4);
    float* biasD1 = (float*)alloc((size_t)G4H * 4);
    float* meanB  = (float*)alloc((size_t)BB * FF * 4);
    float* stdB   = (float*)alloc((size_t)BB * FF * 4);
    float* cL0    = (float*)alloc((size_t)BB * HH * 4);
    float* cL1    = (float*)alloc((size_t)BB * HH * 4);
    float* partial = (float*)alloc((size_t)16 * BB * G4H * 4);   // 16 split-K slots

    const int T = 256;
    auto g1 = [](int n, int t) { return (n + t - 1) / t; };

    // ---- weight conversion (fp32 -> bf16) ----
    k_f32_to_bf16<<<g1(G4H*FF, T), T, 0, stream>>>(enc_Wih0, wEWih0, G4H*FF);
    k_f32_to_bf16<<<g1(G4H*HH, T), T, 0, stream>>>(enc_Whh0, wEWhh0, G4H*HH);
    k_f32_to_bf16<<<g1(G4H*HH, T), T, 0, stream>>>(enc_Wih1, wEWih1, G4H*HH);
    k_f32_to_bf16<<<g1(G4H*HH, T), T, 0, stream>>>(enc_Whh1, wEWhh1, G4H*HH);
    k_f32_to_bf16<<<g1(G4H*FF, T), T, 0, stream>>>(dec_Wih0, wDWih0, G4H*FF);
    k_f32_to_bf16<<<g1(G4H*HH, T), T, 0, stream>>>(dec_Whh0, wDWhh0, G4H*HH);
    k_f32_to_bf16<<<g1(G4H*HH, T), T, 0, stream>>>(dec_Wih1, wDWih1, G4H*HH);
    k_f32_to_bf16<<<g1(G4H*HH, T), T, 0, stream>>>(dec_Whh1, wDWhh1, G4H*HH);
    k_f32_to_bf16<<<g1(FF*HH,  T), T, 0, stream>>>(fc_W,     wFc,    FF*HH);

    k_bias_combine<<<g1(G4H, T), T, 0, stream>>>(enc_bih0, enc_bhh0, biasE0, G4H);
    k_bias_combine<<<g1(G4H, T), T, 0, stream>>>(enc_bih1, enc_bhh1, biasE1, G4H);
    k_bias_combine<<<g1(G4H, T), T, 0, stream>>>(dec_bih0, dec_bhh0, biasD0, G4H);
    k_bias_combine<<<g1(G4H, T), T, 0, stream>>>(dec_bih1, dec_bhh1, biasD1, G4H);

    // ---- zero initial state ----
    k_zero<<<g1(BB*HH, T), T, 0, stream>>>((uint32_t*)cL0, BB*HH);
    k_zero<<<g1(BB*HH, T), T, 0, stream>>>((uint32_t*)cL1, BB*HH);
    k_zero<<<g1(BB*HH/2, T), T, 0, stream>>>((uint32_t*)hL0, BB*HH/2);
    k_zero<<<g1(BB*HH/2, T), T, 0, stream>>>((uint32_t*)hL1, BB*HH/2);
    k_zero<<<g1(BB*FF/2, T), T, 0, stream>>>((uint32_t*)xprev, BB*FF/2);

    // ---- instance norm ----
    k_instnorm<<<g1(BB*FF*32, T), T, 0, stream>>>(x_enc, xn, meanB, stdB);

    const int SH = 8;   // split-K slots for the H-path (K=1024 -> chunks of 128)
    const int PW_BLK = g1(BB * HH, T);

    // ---- encoder: 512 sequential steps, 2 layers each ----
    for (int t = 0; t < SS; ++t) {
        const unsigned short* xt = xn + (size_t)t * BB * FF;
        // layer 0: h-path K=1024 (8 slots) + x-path K=64 (1 slot)
        k_gates_gemm<<<32 * (SH + 1), T, 0, stream>>>(
            hL0, wEWhh0, HH, SH, xt, wEWih0, FF, 1, partial);
        k_lstm_pointwise<<<PW_BLK, T, 0, stream>>>(partial, SH + 1, biasE0, cL0, hL0, nullptr);
        // layer 1: h-path K=1024 (8 slots) + x-path K=1024 (8 slots)
        k_gates_gemm<<<32 * (SH + SH), T, 0, stream>>>(
            hL1, wEWhh1, HH, SH, hL0, wEWih1, HH, SH, partial);
        k_lstm_pointwise<<<PW_BLK, T, 0, stream>>>(partial, SH + SH, biasE1, cL1, hL1, nullptr);
    }

    // ---- decoder: 512 autoregressive steps (states continue from encoder) ----
    for (int t = 0; t < SS; ++t) {
        k_gates_gemm<<<32 * (SH + 1), T, 0, stream>>>(
            hL0, wDWhh0, HH, SH, xprev, wDWih0, FF, 1, partial);
        k_lstm_pointwise<<<PW_BLK, T, 0, stream>>>(partial, SH + 1, biasD0, cL0, hL0, nullptr);
        k_gates_gemm<<<32 * (SH + SH), T, 0, stream>>>(
            hL1, wDWhh1, HH, SH, hL0, wDWih1, HH, SH, partial);
        k_lstm_pointwise<<<PW_BLK, T, 0, stream>>>(partial, SH + SH, biasD1, cL1, hL1, hrelu);
        k_fc<<<2, T, 0, stream>>>(hrelu, wFc, fc_b, meanB, stdB, xprev, out, t);
    }
}
